// MambaModel_10763188044445
// MI455X (gfx1250) — compile-verified, matrix operand order checked by reading, not run on hardware
//
#include <hip/hip_runtime.h>

// ---------------- dims (compile-time, match reference) ----------------
#define NL      2
#define DM      256
#define DIN     512
#define HEADDIM 64
#define NHEADS  8
#define DSTATE  64
#define DCONV   4
#define DIP     1160   // 2*DIN + 2*DSTATE + NHEADS
#define DIPP    1216   // DIP padded to multiple of 64
#define CDIM    640    // DIN + 2*DSTATE
#define BSZ     2
#define SEQL    2048
#define BL      (BSZ * SEQL)   // 4096 rows
#define DH      (4 * DM)       // 1024 MLP hidden

typedef float  v8f   __attribute__((ext_vector_type(8)));
typedef __bf16 v16bf __attribute__((ext_vector_type(16)));

__device__ __forceinline__ __bf16 f2bf(float f) {
    union { float f; unsigned u; } a; a.f = f;
    unsigned r = a.u + 0x7FFFu + ((a.u >> 16) & 1u);   // round-to-nearest-even
    union { unsigned short s; __bf16 b; } o; o.s = (unsigned short)(r >> 16);
    return o.b;
}
__device__ __forceinline__ float bf2f(__bf16 b) {
    union { unsigned short s; __bf16 b; } i; i.b = b;
    union { unsigned u; float f; } o; o.u = (unsigned)i.s << 16;
    return o.f;
}
__device__ __forceinline__ float siluf_(float x) { return x / (1.f + __expf(-x)); }
__device__ __forceinline__ float geluf_(float x) { return 0.5f * x * (1.f + erff(x * 0.70710678118654752f)); }

__device__ __forceinline__ void store_val(float*  p, float v) { *p = v; }
__device__ __forceinline__ void store_val(__bf16* p, float v) { *p = f2bf(v); }

// ---------------- WMMA GEMM with async global->LDS double-buffered staging ----------------
// out[m,n] = act(sum_k A[m,k]*W[n,k] + bias[n]) + res[m,n]
// A: M x K bf16 row-major (M mult of 64, K mult of 32); W: Npad x K bf16 row-major (Npad mult of 64,
// rows >= N zero-filled). 256 threads = 8 wave32 waves; 64x64x32 tile; f32 accumulate.
template <typename OutT>
__global__ __launch_bounds__(256)
void gemm_bf16_wmma_async_kernel(const __bf16* __restrict__ A, const __bf16* __restrict__ W,
                                 const float* __restrict__ bias, const float* __restrict__ residual,
                                 OutT* __restrict__ out, int N, int K, int act)
{
    constexpr int LDSK = 40;  // 32 bf16 + 8 bf16 pad (80B rows; 16B chunks stay aligned)
    __shared__ __bf16 As[2][64 * LDSK];
    __shared__ __bf16 Ws[2][64 * LDSK];

    const int tid     = threadIdx.x;
    const int lane    = tid & 31;        // wave32
    const int wave    = tid >> 5;
    const int lane_lo = lane & 15;
    const int sel     = lane >> 4;       // 0 or 1
    const int mi0     = (2 * wave) & 3;  // wave covers C-subtiles (mi0,ni),(mi0+1,ni)
    const int ni      = wave >> 1;
    const int m0      = blockIdx.y * 64;
    const int n0      = blockIdx.x * 64;
    const int sr      = tid >> 2;        // staging row 0..63
    const int sc8     = (tid & 3) * 8;   // staging k-offset {0,8,16,24} (16B chunks)

    const __bf16* gA = A + (size_t)(m0 + sr) * K + sc8;
    const __bf16* gW = W + (size_t)(n0 + sr) * K + sc8;

    v8f acc0, acc1;
#pragma unroll
    for (int i = 0; i < 8; ++i) { acc0[i] = 0.f; acc1[i] = 0.f; }

    const int nk = K >> 5;

    // one async B128 per matrix per stage per lane: 64x32 bf16 tile = 256 lanes * 16B
    auto issue = [&](int s) {
        const int buf = s & 1;
        const unsigned la = (unsigned)(size_t)&As[buf][sr * LDSK + sc8];
        const unsigned lw = (unsigned)(size_t)&Ws[buf][sr * LDSK + sc8];
        const __bf16* pa = gA + s * 32;
        const __bf16* pw = gW + s * 32;
        asm volatile("global_load_async_to_lds_b128 %0, %1, off" :: "v"(la), "v"(pa) : "memory");
        asm volatile("global_load_async_to_lds_b128 %0, %1, off" :: "v"(lw), "v"(pw) : "memory");
    };

    issue(0);
    for (int s = 0; s < nk; ++s) {
        if (s + 1 < nk) {
            issue(s + 1);
            // async loads retire in order: <=2 outstanding => stage s's 2 transfers are done
            asm volatile("s_wait_asynccnt 0x2" ::: "memory");
        } else {
            asm volatile("s_wait_asynccnt 0x0" ::: "memory");
        }
        __syncthreads();  // make all waves' stage-s transfers visible

        const int buf = s & 1;
        const int kb  = sel * 8;
        // fragment layout (ISA 7.12.2, 16-bit 16x32): lanes 0-15 -> K 0-7 & 16-23,
        // lanes 16-31 -> K 8-15 & 24-31; per-lane row = lane%16.
        auto frag = [&](const __bf16* Ls, int row) {
            const __bf16* p = Ls + row * LDSK + kb;
            v16bf f;
#pragma unroll
            for (int j = 0; j < 8; ++j) { f[j] = p[j]; f[j + 8] = p[j + 16]; }
            return f;
        };
        v16bf a0 = frag(As[buf], (mi0 + 0) * 16 + lane_lo);
        v16bf a1 = frag(As[buf], (mi0 + 1) * 16 + lane_lo);
        v16bf w0 = frag(Ws[buf], ni * 16 + lane_lo);

        acc0 = __builtin_amdgcn_wmma_f32_16x16x32_bf16(false, a0, false, w0, (short)0, acc0, false, false);
        acc1 = __builtin_amdgcn_wmma_f32_16x16x32_bf16(false, a1, false, w0, (short)0, acc1, false, false);

        __syncthreads();  // all waves done reading buf before it is re-issued (stage s+2)
    }

    // C layout: VGPR r, lanes 0-15 -> M=r, lanes 16-31 -> M=8+r; N = lane%16.
    const int n = n0 + ni * 16 + lane_lo;
    if (n < N) {
        const float bv = bias ? bias[n] : 0.f;
#pragma unroll
        for (int st = 0; st < 2; ++st) {
            const v8f acc   = st ? acc1 : acc0;
            const int mbase = m0 + (mi0 + st) * 16 + sel * 8;
#pragma unroll
            for (int r = 0; r < 8; ++r) {
                float v = acc[r] + bv;
                if (act) v = geluf_(v);
                const size_t off = (size_t)(mbase + r) * N + n;
                if (residual) v += residual[off];
                store_val(out + off, v);
            }
        }
    }
}

// ---------------- f32 -> bf16 weight conversion with zero row-padding ----------------
__global__ __launch_bounds__(256)
void f32_to_bf16_pad_kernel(const float* __restrict__ src, __bf16* __restrict__ dst,
                            int rows, int cols, int rowsPad)
{
    const int idx = blockIdx.x * 256 + threadIdx.x;
    if (idx >= rowsPad * cols) return;
    const int r = idx / cols;
    dst[idx] = (r < rows) ? f2bf(src[idx]) : f2bf(0.f);
}

// ---------------- LayerNorm over DM=256 -> bf16, one row per 256-thread block ----------------
__global__ __launch_bounds__(256)
void layernorm_kernel(const float* __restrict__ x, const float* __restrict__ w,
                      const float* __restrict__ b, __bf16* __restrict__ out)
{
    __shared__ float red[256];
    const int row = blockIdx.x, t = threadIdx.x;
    const float v = x[(size_t)row * DM + t];
    red[t] = v; __syncthreads();
#pragma unroll
    for (int s = 128; s > 0; s >>= 1) { if (t < s) red[t] += red[t + s]; __syncthreads(); }
    const float mu = red[0] * (1.f / DM);
    __syncthreads();
    const float d = v - mu;
    red[t] = d * d; __syncthreads();
#pragma unroll
    for (int s = 128; s > 0; s >>= 1) { if (t < s) red[t] += red[t + s]; __syncthreads(); }
    const float var = red[0] * (1.f / DM);
    out[(size_t)row * DM + t] = f2bf(d * rsqrtf(var + 1e-5f) * w[t] + b[t]);
}

// ---------------- dt = softplus(zx[..., DIN+CDIM+h] + dt_bias[h]) ----------------
__global__ __launch_bounds__(256)
void dt_softplus_kernel(const float* __restrict__ zx, const float* __restrict__ dtb,
                        float* __restrict__ dts)
{
    const int idx = blockIdx.x * 256 + threadIdx.x;
    if (idx >= BL * NHEADS) return;
    const int row = idx / NHEADS, h = idx % NHEADS;
    const float v = zx[(size_t)row * DIP + DIN + CDIM + h] + dtb[h];
    dts[idx] = (v > 20.f) ? v : log1pf(__expf(v));
}

// ---------------- causal depthwise conv1d (K=4) + bias + SiLU ----------------
__global__ __launch_bounds__(256)
void conv_silu_kernel(const float* __restrict__ zx, const float* __restrict__ cw,
                      const float* __restrict__ cb, float* __restrict__ xc)
{
    const int idx = blockIdx.x * 256 + threadIdx.x;
    if (idx >= BL * CDIM) return;
    const int c  = idx % CDIM;
    const int bt = idx / CDIM;
    const int t  = bt % SEQL;
    const int b  = bt / SEQL;
    float acc = cb[c];
    const float* w = cw + (size_t)c * DCONV;
#pragma unroll
    for (int j = 0; j < DCONV; ++j) {
        const int tt = t + j - (DCONV - 1);
        if (tt >= 0) acc += w[j] * zx[(size_t)(b * SEQL + tt) * DIP + DIN + c];
    }
    xc[idx] = siluf_(acc);
}

// ---------------- sequential SSM scan: 1 block per (batch, head) ----------------
// thread t owns state h[p = t%64][n0 + 0..15]; B/C/x staged in LDS per step.
__global__ __launch_bounds__(256)
void scan_kernel(const float* __restrict__ xc, const float* __restrict__ dts,
                 const float* __restrict__ A_log, const float* __restrict__ Dp,
                 float* __restrict__ y)
{
    __shared__ float xs[HEADDIM], Bsh[DSTATE], Csh[DSTATE], yp[256];
    const int bh = blockIdx.x;
    const int b  = bh / NHEADS, h = bh % NHEADS;
    const int t  = threadIdx.x;
    const int p  = t & 63;
    const int n0 = (t >> 6) * 16;
    const float A  = -__expf(A_log[h]);
    const float Dv = Dp[h];
    float hreg[16];
#pragma unroll
    for (int j = 0; j < 16; ++j) hreg[j] = 0.f;

    for (int step = 0; step < SEQL; ++step) {
        const size_t rb = (size_t)(b * SEQL + step) * CDIM;
        if (t < 64)       xs[t]        = xc[rb + h * HEADDIM + t];
        else if (t < 128) Bsh[t - 64]  = xc[rb + DIN + (t - 64)];
        else if (t < 192) Csh[t - 128] = xc[rb + DIN + DSTATE + (t - 128)];
        __syncthreads();

        const float dtt = dts[(size_t)(b * SEQL + step) * NHEADS + h];
        const float dA  = __expf(dtt * A);
        const float dtx = dtt * xs[p];
        float acc = 0.f;
#pragma unroll
        for (int j = 0; j < 16; ++j) {
            hreg[j] = dA * hreg[j] + dtx * Bsh[n0 + j];
            acc += hreg[j] * Csh[n0 + j];
        }
        yp[t] = acc;
        __syncthreads();
        if (t < 64) {
            const float s = yp[t] + yp[t + 64] + yp[t + 128] + yp[t + 192];
            y[(size_t)(b * SEQL + step) * DIN + h * HEADDIM + t] = s + Dv * xs[t];
        }
        __syncthreads();
    }
}

// ---------------- gated RMSNorm (dim 512) -> bf16, z from zxbcdt ----------------
__global__ __launch_bounds__(256)
void gated_rmsnorm_kernel(const float* __restrict__ y, const float* __restrict__ zx,
                          const float* __restrict__ rw, __bf16* __restrict__ out)
{
    __shared__ float red[256];
    const int row = blockIdx.x, t = threadIdx.x;
    const float* z  = zx + (size_t)row * DIP;
    const float* yr = y + (size_t)row * DIN;
    __bf16* o = out + (size_t)row * DIN;
    const float g0 = yr[t]       * siluf_(z[t]);
    const float g1 = yr[t + 256] * siluf_(z[t + 256]);
    red[t] = g0 * g0 + g1 * g1;
    __syncthreads();
#pragma unroll
    for (int s = 128; s > 0; s >>= 1) { if (t < s) red[t] += red[t + s]; __syncthreads(); }
    const float scale = rsqrtf(red[0] * (1.f / DIN) + 1e-5f);
    o[t]       = f2bf(g0 * scale * rw[t]);
    o[t + 256] = f2bf(g1 * scale * rw[t + 256]);
}

// ---------------- elementwise helpers ----------------
__global__ __launch_bounds__(256)
void copy_kernel(const float* __restrict__ in, float* __restrict__ out, int n)
{
    const int i = blockIdx.x * 256 + threadIdx.x;
    if (i < n) out[i] = in[i];
}
__global__ __launch_bounds__(256)
void gelu_bf16_kernel(const float* __restrict__ in, __bf16* __restrict__ out, int n)
{
    const int i = blockIdx.x * 256 + threadIdx.x;
    if (i < n) out[i] = f2bf(geluf_(in[i]));
}

// ---------------- orchestration ----------------
extern "C" void kernel_launch(void* const* d_in, const int* in_sizes, int n_in,
                              void* d_out, int out_size, void* d_ws, size_t ws_size,
                              hipStream_t stream)
{
    (void)in_sizes; (void)n_in; (void)out_size; (void)ws_size;
    const float* frame  = (const float*)d_in[0];
    const float* ln1_w  = (const float*)d_in[1];
    const float* ln1_b  = (const float*)d_in[2];
    const float* in_w   = (const float*)d_in[3];   // (NL, DIP, DM)
    const float* conv_w = (const float*)d_in[4];   // (NL, CDIM, DCONV)
    const float* conv_b = (const float*)d_in[5];
    const float* dt_b   = (const float*)d_in[6];   // (NL, NHEADS)
    const float* A_log  = (const float*)d_in[7];
    const float* Dp     = (const float*)d_in[8];
    const float* rms_w  = (const float*)d_in[9];   // (NL, DIN)
    const float* out_w  = (const float*)d_in[10];  // (NL, DM, DIN)
    const float* ln2_w  = (const float*)d_in[11];
    const float* ln2_b  = (const float*)d_in[12];
    const float* w1     = (const float*)d_in[13];  // (NL, DH, DM)
    const float* b1     = (const float*)d_in[14];
    const float* w2     = (const float*)d_in[15];  // (NL, DM, DH)
    const float* b2     = (const float*)d_in[16];
    const float* fl_w   = (const float*)d_in[17];  // (DM, DM)
    const float* fl_b   = (const float*)d_in[18];
    float* outp = (float*)d_out;

    char* ws = (char*)d_ws;
    size_t off = 0;
    auto take = [&](size_t bytes) {
        void* p = ws + off;
        off += (bytes + 255) & ~(size_t)255;
        return p;
    };
    float*  XRES  = (float*)take((size_t)BL * DM * 4);        // running residual x (f32)
    float*  ZX    = (float*)take((size_t)BL * DIP * 4);       // zxbcdt (f32)
    float*  XC    = (float*)take((size_t)BL * CDIM * 4);      // conv output (f32)
    float*  DTS   = (float*)take((size_t)BL * NHEADS * 4);    // softplus(dt)
    float*  Y     = (float*)take((size_t)BL * DIN * 4);       // SSM output (f32)
    __bf16* HBF   = (__bf16*)take((size_t)BL * DM * 2);       // LN / gelu output (bf16)
    __bf16* YBF   = (__bf16*)take((size_t)BL * DIN * 2);      // gated-normed (bf16)
    __bf16* HIDBF = (__bf16*)take((size_t)BL * DH * 2);       // MLP hidden (bf16)
    __bf16* WBin  = (__bf16*)take((size_t)NL * DIPP * DM * 2);
    __bf16* WBout = (__bf16*)take((size_t)NL * DM * DIN * 2);
    __bf16* WB1   = (__bf16*)take((size_t)NL * DH * DM * 2);
    __bf16* WB2   = (__bf16*)take((size_t)NL * DM * DH * 2);
    __bf16* WBfl  = (__bf16*)take((size_t)DM * DM * 2);

    auto cvt = [&](const float* s, __bf16* d, int rows, int cols, int rowsPad) {
        const int total = rowsPad * cols;
        f32_to_bf16_pad_kernel<<<(total + 255) / 256, 256, 0, stream>>>(s, d, rows, cols, rowsPad);
    };
    for (int i = 0; i < NL; ++i) {
        cvt(in_w  + (size_t)i * DIP * DM, WBin  + (size_t)i * DIPP * DM, DIP, DM, DIPP);
        cvt(out_w + (size_t)i * DM * DIN, WBout + (size_t)i * DM * DIN,  DM,  DIN, DM);
        cvt(w1    + (size_t)i * DH * DM,  WB1   + (size_t)i * DH * DM,   DH,  DM,  DH);
        cvt(w2    + (size_t)i * DM * DH,  WB2   + (size_t)i * DM * DH,   DM,  DH,  DM);
    }
    cvt(fl_w, WBfl, DM, DM, DM);

    copy_kernel<<<(BL * DM) / 256, 256, 0, stream>>>(frame, XRES, BL * DM);

    const dim3 gIn((DIP + 63) / 64, BL / 64);
    const dim3 gDm(DM / 64, BL / 64);
    const dim3 gH(DH / 64, BL / 64);

    for (int i = 0; i < NL; ++i) {
        layernorm_kernel<<<BL, 256, 0, stream>>>(XRES, ln1_w + i * DM, ln1_b + i * DM, HBF);
        gemm_bf16_wmma_async_kernel<float><<<gIn, 256, 0, stream>>>(
            HBF, WBin + (size_t)i * DIPP * DM, nullptr, nullptr, ZX, DIP, DM, 0);
        dt_softplus_kernel<<<(BL * NHEADS) / 256, 256, 0, stream>>>(ZX, dt_b + i * NHEADS, DTS);
        conv_silu_kernel<<<(BL * CDIM) / 256, 256, 0, stream>>>(
            ZX, conv_w + (size_t)i * CDIM * DCONV, conv_b + i * CDIM, XC);
        scan_kernel<<<BSZ * NHEADS, 256, 0, stream>>>(XC, DTS, A_log + i * NHEADS, Dp + i * NHEADS, Y);
        gated_rmsnorm_kernel<<<BL, 256, 0, stream>>>(Y, ZX, rms_w + (size_t)i * DIN, YBF);
        gemm_bf16_wmma_async_kernel<float><<<gDm, 256, 0, stream>>>(
            YBF, WBout + (size_t)i * DM * DIN, nullptr, XRES, XRES, DM, DIN, 0);
        layernorm_kernel<<<BL, 256, 0, stream>>>(XRES, ln2_w + i * DM, ln2_b + i * DM, HBF);
        gemm_bf16_wmma_async_kernel<__bf16><<<gH, 256, 0, stream>>>(
            HBF, WB1 + (size_t)i * DH * DM, b1 + i * DH, nullptr, HIDBF, DH, DM, 1);
        gemm_bf16_wmma_async_kernel<float><<<gDm, 256, 0, stream>>>(
            HIDBF, WB2 + (size_t)i * DM * DH, b2 + i * DM, XRES, XRES, DM, DH, 0);
    }

    gelu_bf16_kernel<<<(BL * DM) / 256, 256, 0, stream>>>(XRES, HBF, BL * DM);
    gemm_bf16_wmma_async_kernel<float><<<gDm, 256, 0, stream>>>(
        HBF, WBfl, fl_b, frame, outp, DM, DM, 0);
}
